// SparseConv2d_3719441678734
// MI455X (gfx1250) — matrix operand order repaired and checked by reading, not compiled, not run on hardware
//
#include <hip/hip_runtime.h>
#include <hip/hip_bf16.h>

// ---------------------------------------------------------------------------
// 3x3 conv, N=16, Cin=Cout=64, H=W=256, pad=1, f32 in/out.
// Implicit GEMM on CDNA5 WMMA (bf16, f32 accum), bf16x2 weight split:
//   y = Ah*Bh + Ah*Bl   (A = x in bf16, B = w split hi+lo bf16)
// WG = 4 waves = one full 256-px row, TWO output rows per WG. Wave tile:
// 64 px x 64 co (4x4 register blocking, 16 v8f accumulators). p-loop rolled;
// B-fragment region lives at LDS offset 0 so ALL fragment loads fold into
// 16-bit DS immediate offsets (no per-load address VALU, no WAR-hazard nops).
// ---------------------------------------------------------------------------

typedef __attribute__((ext_vector_type(16))) __bf16 v16bf;
typedef __attribute__((ext_vector_type(8)))  float  v8f;
typedef __attribute__((ext_vector_type(4)))  float  vf4;

union Frag32 { uint4 q[2]; v16bf v; };

// bf16 round-to-nearest-even from f32
__device__ __forceinline__ unsigned short f2bf(float f) {
    unsigned u = __float_as_uint(f);
    u += 0x7FFFu + ((u >> 16) & 1u);
    return (unsigned short)(u >> 16);
}
__device__ __forceinline__ float bf2f(unsigned short h) {
    return __uint_as_float(((unsigned)h) << 16);
}

// ---------------------------------------------------------------------------
#define CIN   64
#define COUT  64
#define HH    256
#define WW    256
#define NB    16

#define RSH       72                   // halves per (y,w) row (64 data + 8 pad)
#define XW        258                  // w extent incl. halo
#define NYROW     4                    // input rows staged (2 out rows + halo)
#define NFRAG     144                  // 9 pos * 2 kblk * 4 ntile * 2 part
#define B_BYTES   (NFRAG*1024)         // 147456  (B region at LDS offset 0)
#define OFF_X     B_BYTES              // 147456  (x-tile region)
#define X_BYTES   (NYROW*XW*RSH*2)     // 148608
#define OFF_O     (OFF_X + X_BYTES)    // 296064  (rotating out buffer)
#define OST_STRIDE 20                  // floats per co row in rotating out buf
#define SMEM_TOTAL (OFF_O + 4*COUT*OST_STRIDE*4)   // 316544 < 320KB

// ---------------------------------------------------------------------------
// Kernel 1: pack weights (OIHW f32) into per-lane B-fragment layout, bf16 hi/lo.
// dword g = ((f*2 + plane)*32 + lane)*4 + j
//   f = p*16 + kblk*8 + ntile*2 + part     (p = kh*3+kw)  -> p-major, 16KB/p
//   B 32x16 bf16: lane<16 -> K=0..7,16..23, lane>=16 -> K=8..15,24..31 (table order)
// ---------------------------------------------------------------------------
__global__ void pack_weights_kernel(const float* __restrict__ wt,
                                    unsigned int* __restrict__ frag) {
    for (int g = threadIdx.x + blockIdx.x * blockDim.x; g < NFRAG * 256;
         g += blockDim.x * gridDim.x) {
        int j     = g & 3;
        int lane  = (g >> 2) & 31;
        int pl    = (g >> 7) & 1;
        int f     = g >> 8;
        int part  = f & 1;
        int ntile = (f >> 1) & 3;
        int kblk  = (f >> 3) & 1;
        int p     = f >> 4;
        int v     = pl * 4 + j;
        int kb    = ((v >> 2) << 4) + ((lane >> 4) << 3) + (v & 3) * 2;
        int co    = ntile * 16 + (lane & 15);
        int kh    = p / 3, kw = p % 3;
        unsigned int d = 0;
        #pragma unroll
        for (int t = 0; t < 2; ++t) {
            int ci = kblk * 32 + kb + t;
            float w = wt[((co * CIN + ci) * 3 + kh) * 3 + kw];
            unsigned short hi = f2bf(w);
            unsigned short o16 = (part == 0) ? hi : f2bf(w - bf2f(hi));
            d |= ((unsigned int)o16) << (16 * t);
        }
        frag[g] = d;
    }
}

// ---------------------------------------------------------------------------
// Kernel 2: conv. Grid = (128 row-pairs, 16 n), block = 128 (4 waves).
// ---------------------------------------------------------------------------
__global__ void __launch_bounds__(128, 1)
conv3x3_wmma_kernel(const float* __restrict__ x,
                    const unsigned int* __restrict__ wfrag,
                    const float* __restrict__ bias,
                    float* __restrict__ out) {
    extern __shared__ unsigned char smem[];
    unsigned short* xs = (unsigned short*)(smem + OFF_X);  // [y][w][ci]

    const int tid = threadIdx.x;
    const int h   = blockIdx.x * 2;               // first of two output rows
    const int n   = blockIdx.y;

    // ---- Stage B fragments (hi+lo) ws -> LDS (region at offset 0) ----
    {
        const uint4* src = (const uint4*)wfrag;
        uint4*       dst = (uint4*)smem;
        for (int i = tid; i < B_BYTES / 16; i += 128) dst[i] = src[i];
    }
    // ---- Constant-zero halo columns (w=0 and w=257) ----
    for (int i = tid; i < NYROW * CIN * 2; i += 128) {     // 512
        int y  = i >> 7;
        int ci = (i >> 1) & 63;
        int w  = (i & 1) ? (XW - 1) : 0;
        xs[(y * XW + w) * RSH + ci] = 0;
    }
    // ---- Stage x tile -> LDS transposed to [y][w][ci]; float4 reads,
    //      pure shift/mask index decomposition (no integer divides) ----
    for (int idx = tid; idx < NYROW * CIN * 64; idx += 128) {  // 16384 float4s
        int y  = idx >> 12;          // / (64*64)
        int ci = (idx >> 6) & 63;
        int c  = idx & 63;           // 4-wide chunk along w
        int gy = h + y - 1;
        float4 v = make_float4(0.f, 0.f, 0.f, 0.f);
        if ((unsigned)gy < (unsigned)HH)
            v = *(const float4*)(x + (((size_t)n * CIN + ci) * HH + gy) * WW + c * 4);
        int base = (y * XW + (c * 4 + 1)) * RSH + ci;   // lds w = gw + 1
        xs[base          ] = f2bf(v.x);
        xs[base + RSH    ] = f2bf(v.y);
        xs[base + 2 * RSH] = f2bf(v.z);
        xs[base + 3 * RSH] = f2bf(v.w);
    }
    __syncthreads();

    const int wave = tid >> 5;
    const int lane = tid & 31;
    const int M    = lane & 15;   // A row (pixel) within 16-tile
    const int lh   = lane >> 4;   // half-wave select

    // Per-lane base pointers; all fragment loads below use compile-time
    // immediate DS offsets relative to one per-iteration base.
    const unsigned short* arow = xs + (wave * 64 + M) * RSH + lh * 8;
    const char* bP = (const char*)smem + lane * 16;     // B region @ offset 0

    float* ost = (float*)(smem + OFF_O) + wave * (COUT * OST_STRIDE);

    float bv[4];
    #pragma unroll
    for (int nt = 0; nt < 4; ++nt) bv[nt] = bias[nt * 16 + M];

    #pragma unroll 1
    for (int h2 = 0; h2 < 2; ++h2) {
        v8f acc[4][4];                // [ntile][atile]
        #pragma unroll
        for (int i = 0; i < 4; ++i)
            #pragma unroll
            for (int j = 0; j < 4; ++j)
                acc[i][j] = (v8f){0, 0, 0, 0, 0, 0, 0, 0};

        #pragma unroll 1
        for (int p = 0; p < 9; ++p) {
            const int kh = p / 3, kw = p - kh * 3;
            // one A base (offsets: a*2304 + kblk*64 + {0,32..} bytes, < 8KB)
            const unsigned short* ap = arow + ((kh + h2) * XW + kw) * RSH;
            // one B base (offsets: local_frag*1024 + plane*512, all < 16KB)
            const char* bp = bP + p * 16384;

            Frag32 A[4][2];           // [atile][kblk]
            #pragma unroll
            for (int kblk = 0; kblk < 2; ++kblk)
                #pragma unroll
                for (int a = 0; a < 4; ++a) {
                    A[a][kblk].q[0] = *(const uint4*)(ap + a * 1152 + kblk * 32);
                    A[a][kblk].q[1] = *(const uint4*)(ap + a * 1152 + kblk * 32 + 16);
                }
            #pragma unroll
            for (int kblk = 0; kblk < 2; ++kblk)
                #pragma unroll
                for (int nt = 0; nt < 4; ++nt) {
                    const int lf = kblk * 8 + nt * 2;   // local frag (hi)
                    Frag32 Bh, Bl;
                    Bh.q[0] = *(const uint4*)(bp + lf * 1024);
                    Bh.q[1] = *(const uint4*)(bp + lf * 1024 + 512);
                    Bl.q[0] = *(const uint4*)(bp + lf * 1024 + 1024);
                    Bl.q[1] = *(const uint4*)(bp + lf * 1024 + 1536);
                    #pragma unroll
                    for (int a = 0; a < 4; ++a) {
                        acc[nt][a] = __builtin_amdgcn_wmma_f32_16x16x32_bf16(
                            false, A[a][kblk].v, false, Bh.v, (short)0,
                            acc[nt][a], false, false);
                        acc[nt][a] = __builtin_amdgcn_wmma_f32_16x16x32_bf16(
                            false, A[a][kblk].v, false, Bl.v, (short)0,
                            acc[nt][a], false, false);
                    }
                }
        }

        // ---- bias + store via per-wave rotating LDS buffer (one atile/round)
        //      LDS ops within a wave stay in order; no barriers needed. ----
        #pragma unroll 1
        for (int a = 0; a < 4; ++a) {
            #pragma unroll
            for (int nt = 0; nt < 4; ++nt) {
                const int co = nt * 16 + M;
                #pragma unroll
                for (int r = 0; r < 8; ++r)
                    ost[co * OST_STRIDE + r + 8 * lh] = acc[nt][a][r] + bv[nt];
            }
            #pragma unroll
            for (int it = 0; it < 8; ++it) {
                const int idx = it * 32 + lane;   // 256 float4 chunks
                const int co  = idx >> 2;
                const int q   = (idx & 3) * 4;
                const float* sp = ost + co * OST_STRIDE + q;
                vf4 vv = {sp[0], sp[1], sp[2], sp[3]};
                __builtin_nontemporal_store(vv,
                    (vf4*)(out + (((size_t)n * COUT + co) * HH + (h + h2)) * WW
                           + wave * 64 + a * 16 + q));
            }
        }
    }
}

// ---------------------------------------------------------------------------
extern "C" void kernel_launch(void* const* d_in, const int* in_sizes, int n_in,
                              void* d_out, int out_size, void* d_ws, size_t ws_size,
                              hipStream_t stream) {
    (void)in_sizes; (void)n_in; (void)out_size; (void)ws_size;
    const float* x    = (const float*)d_in[0];
    const float* wt   = (const float*)d_in[1];
    const float* bias = (const float*)d_in[2];
    float*       out  = (float*)d_out;
    unsigned int* frag = (unsigned int*)d_ws;   // 147456 bytes used

    pack_weights_kernel<<<dim3(16), 256, 0, stream>>>(wt, frag);
    conv3x3_wmma_kernel<<<dim3(HH / 2, NB), 128, SMEM_TOTAL, stream>>>(x, frag, bias, out);
}
